// TransformerEncoderBlock_11012296147294
// MI455X (gfx1250) — matrix operand order repaired and checked by reading, not compiled, not run on hardware
//
#include <hip/hip_runtime.h>
#include <stdint.h>

// ---------- types ----------
typedef __attribute__((ext_vector_type(16))) __bf16 v16bf;
typedef __attribute__((ext_vector_type(8)))  float  v8f;

__device__ __forceinline__ uint16_t f2bf(float f) {
  uint32_t u = __float_as_uint(f);
  return (uint16_t)((u + 0x7FFFu + ((u >> 16) & 1u)) >> 16);
}

union FragB16 { uint32_t u[8]; v16bf v; };

// LDS tile strides (elements). 40*2B=80B row stride (16B aligned), 136*2B=272B.
#define LDSA 40
#define LDSB 136

// ---------------------------------------------------------------------------
// Tiled bf16 WMMA GEMM: C[M,N] = A[M,K] x (TRANSB ? B[N,K]^T : B[K,N])
// Block = 256 threads = 8 waves; 128x128 tile; each wave 32x64 (2x4 wmma).
// EPI: 0 = f32 store * scale (batched)       1 = bf16 store (batched)
//      2 = f32 store + residual              3 = bf16 store of relu(acc+bias)
//      4 = f32 store acc + bias + residual
// ---------------------------------------------------------------------------
template<bool TRANSB, int EPI>
__launch_bounds__(256)
__global__ void gemm_bf16_wmma(const uint16_t* __restrict__ A,
                               const uint16_t* __restrict__ Bm,
                               void* __restrict__ Cv,
                               int K_, int ldA, int ldB, int ldC,
                               long long aBatch, long long bBatch, long long cBatch,
                               const float* __restrict__ bias,
                               const float* __restrict__ residual,
                               float scale)
{
  __shared__ __align__(16) uint16_t As[128 * LDSA];
  __shared__ __align__(16) uint16_t Bs[128 * LDSA];  // covers both layouts

  const int t    = threadIdx.x;
  const int z    = blockIdx.z;
  A  += (long long)z * aBatch;
  Bm += (long long)z * bBatch;

  const int tileM = blockIdx.y * 128;
  const int tileN = blockIdx.x * 128;

  const int lane = t & 31;
  const int wid  = t >> 5;
  const int mW   = (wid & 3) * 32;   // wave offset in M within tile
  const int nW   = (wid >> 2) * 64;  // wave offset in N within tile

  v8f acc[2][4];
#pragma unroll
  for (int i = 0; i < 2; ++i)
#pragma unroll
    for (int j = 0; j < 4; ++j)
      acc[i][j] = (v8f){0.f, 0.f, 0.f, 0.f, 0.f, 0.f, 0.f, 0.f};

  for (int k0 = 0; k0 < K_; k0 += 32) {
    // ---- A tile: 128 rows x 32 k (row-major in LDS, stride LDSA) ----
#pragma unroll
    for (int i = 0; i < 2; ++i) {
      int v   = t + i * 256;       // 0..511
      int row = v >> 2;            // 0..127
      int col = (v & 3) << 3;      // 0,8,16,24
      const uint16_t* gp = A + (long long)(tileM + row) * ldA + (k0 + col);
      *(uint4*)&As[row * LDSA + col] = *(const uint4*)gp;
    }
    // ---- B tile ----
    if (TRANSB) {  // B stored [N,K]: load as [n][k], same layout as A tile
#pragma unroll
      for (int i = 0; i < 2; ++i) {
        int v   = t + i * 256;
        int row = v >> 2;
        int col = (v & 3) << 3;
        const uint16_t* gp = Bm + (long long)(tileN + row) * ldB + (k0 + col);
        *(uint4*)&Bs[row * LDSA + col] = *(const uint4*)gp;
      }
    } else {       // B stored [K,N]: load as [k][n], stride LDSB
#pragma unroll
      for (int i = 0; i < 2; ++i) {
        int v  = t + i * 256;
        int kr = v >> 4;           // 0..31
        int nc = (v & 15) << 3;    // 0..120
        const uint16_t* gp = Bm + (long long)(k0 + kr) * ldB + (tileN + nc);
        *(uint4*)&Bs[kr * LDSB + nc] = *(const uint4*)gp;
      }
    }
    __syncthreads();

    // ---- fragment loads, contiguous k = 16*(lane>>4) + s for A and B ----
    const int lm   = lane & 15;
    const int hk8  = (lane >> 4) << 3;   // unused; A and B both use hk16
    const int hk16 = (lane >> 4) << 4;   // B: K-half offset 0/16

    FragB16 a[2];
#pragma unroll
    for (int ti = 0; ti < 2; ++ti) {
      const uint16_t* base = &As[(mW + ti * 16 + lm) * LDSA];
#pragma unroll
      for (int i = 0; i < 4; ++i) a[ti].u[i]     = *(const uint32_t*)&base[hk16 + 2 * i];
#pragma unroll
      for (int i = 0; i < 4; ++i) a[ti].u[4 + i] = *(const uint32_t*)&base[hk16 + 8 + 2 * i];
    }
    FragB16 b[4];
#pragma unroll
    for (int tj = 0; tj < 4; ++tj) {
      int nn = nW + tj * 16 + lm;
      if (TRANSB) {
        const uint16_t* base = &Bs[nn * LDSA];
#pragma unroll
        for (int j = 0; j < 8; ++j) b[tj].u[j] = *(const uint32_t*)&base[hk16 + 2 * j];
      } else {
#pragma unroll
        for (int j = 0; j < 8; ++j) {
          int k = hk16 + 2 * j;
          uint32_t lo = Bs[k * LDSB + nn];
          uint32_t hi = Bs[(k + 1) * LDSB + nn];
          b[tj].u[j] = lo | (hi << 16);
        }
      }
    }
#pragma unroll
    for (int ti = 0; ti < 2; ++ti)
#pragma unroll
      for (int tj = 0; tj < 4; ++tj)
        acc[ti][tj] = __builtin_amdgcn_wmma_f32_16x16x32_bf16(
            false, a[ti].v, false, b[tj].v, (short)0, acc[ti][tj], false, false);
    __syncthreads();
  }

  // ---- epilogue (C layout: lane n = lane&15, m = r + 8*(lane>>4)) ----
  const int ln = lane & 15;
  const int lr = (lane >> 4) << 3;
#pragma unroll
  for (int ti = 0; ti < 2; ++ti) {
#pragma unroll
    for (int tj = 0; tj < 4; ++tj) {
      int gn  = tileN + nW + tj * 16 + ln;
      int gm0 = tileM + mW + ti * 16 + lr;
#pragma unroll
      for (int r = 0; r < 8; ++r) {
        long long idx = (long long)(gm0 + r) * ldC + gn;
        float v = acc[ti][tj][r];
        if (EPI == 0) {
          ((float*)Cv)[(long long)z * cBatch + idx] = v * scale;
        } else if (EPI == 1) {
          ((uint16_t*)Cv)[(long long)z * cBatch + idx] = f2bf(v);
        } else if (EPI == 2) {
          ((float*)Cv)[idx] = v + residual[idx];
        } else if (EPI == 3) {
          v += bias[gn];
          ((uint16_t*)Cv)[idx] = f2bf(v > 0.f ? v : 0.f);
        } else {
          ((float*)Cv)[idx] = v + bias[gn] + residual[idx];
        }
      }
    }
  }
}

// ---------------------------------------------------------------------------
// fp32 -> bf16 conversion (vectorized x4)
// ---------------------------------------------------------------------------
__global__ void cvt_bf16_kernel(const float* __restrict__ src,
                                uint16_t* __restrict__ dst, int n4) {
  int i = blockIdx.x * 256 + threadIdx.x;
  if (i < n4) {
    float4 f = ((const float4*)src)[i];
    ushort4 o;
    o.x = f2bf(f.x); o.y = f2bf(f.y); o.z = f2bf(f.z); o.w = f2bf(f.w);
    ((ushort4*)dst)[i] = o;
  }
}

// ---------------------------------------------------------------------------
// Row softmax: fp32 scores -> bf16 probabilities. One block per row.
// ---------------------------------------------------------------------------
__global__ void softmax_rows(const float* __restrict__ S_,
                             uint16_t* __restrict__ P, int cols) {
  const int tid = threadIdx.x;
  const long long row = blockIdx.x;
  const float* sr = S_ + row * cols;
  uint16_t*    pr = P  + row * cols;
  __shared__ float red[256];

  float m = -3.402823466e38f;
  for (int c = tid; c < cols; c += 256) m = fmaxf(m, sr[c]);
  red[tid] = m; __syncthreads();
  for (int s = 128; s > 0; s >>= 1) {
    if (tid < s) red[tid] = fmaxf(red[tid], red[tid + s]);
    __syncthreads();
  }
  const float rowmax = red[0]; __syncthreads();

  float acc = 0.f;
  for (int c = tid; c < cols; c += 256) acc += __expf(sr[c] - rowmax);
  red[tid] = acc; __syncthreads();
  for (int s = 128; s > 0; s >>= 1) {
    if (tid < s) red[tid] += red[tid + s];
    __syncthreads();
  }
  const float inv = 1.f / red[0]; __syncthreads();

  for (int c = tid; c < cols; c += 256)
    pr[c] = f2bf(__expf(sr[c] - rowmax) * inv);
}

// ---------------------------------------------------------------------------
// LayerNorm over (S,M) per batch: hierarchical sum / sumsq reduction
// ---------------------------------------------------------------------------
__global__ void ln_reduce_partial(const float* __restrict__ src,
                                  float* __restrict__ partials,
                                  int perBatch, int parts) {
  const int b = blockIdx.x / parts;
  const int p = blockIdx.x % parts;
  const int chunk = perBatch / parts;
  const float* base = src + (long long)b * perBatch + (long long)p * chunk;
  float s = 0.f, s2 = 0.f;
  for (int i = threadIdx.x; i < chunk; i += 256) {
    float v = base[i];
    s += v; s2 += v * v;
  }
  __shared__ float r1[256], r2[256];
  r1[threadIdx.x] = s; r2[threadIdx.x] = s2; __syncthreads();
  for (int st = 128; st > 0; st >>= 1) {
    if (threadIdx.x < st) {
      r1[threadIdx.x] += r1[threadIdx.x + st];
      r2[threadIdx.x] += r2[threadIdx.x + st];
    }
    __syncthreads();
  }
  if (threadIdx.x == 0) {
    partials[2 * blockIdx.x]     = r1[0];
    partials[2 * blockIdx.x + 1] = r2[0];
  }
}

__global__ void ln_finalize(const float* __restrict__ partials,
                            float* __restrict__ stats, int parts, float invN) {
  const int b = blockIdx.x;
  float s = 0.f, s2 = 0.f;
  for (int p = threadIdx.x; p < parts; p += 256) {
    s  += partials[2 * (b * parts + p)];
    s2 += partials[2 * (b * parts + p) + 1];
  }
  __shared__ float r1[256], r2[256];
  r1[threadIdx.x] = s; r2[threadIdx.x] = s2; __syncthreads();
  for (int st = 128; st > 0; st >>= 1) {
    if (threadIdx.x < st) {
      r1[threadIdx.x] += r1[threadIdx.x + st];
      r2[threadIdx.x] += r2[threadIdx.x + st];
    }
    __syncthreads();
  }
  if (threadIdx.x == 0) {
    float mean = r1[0] * invN;
    float var  = r2[0] * invN - mean * mean;
    stats[2 * b]     = mean;
    stats[2 * b + 1] = rsqrtf(var + 1e-5f);
  }
}

template<bool WB16>
__global__ void ln_apply(const float* __restrict__ src,
                         const float* __restrict__ stats,
                         const float* __restrict__ w,
                         const float* __restrict__ bsh,
                         float* __restrict__ outF,
                         uint16_t* __restrict__ outB, int perBatch) {
  long long idx = (long long)blockIdx.x * 256 + threadIdx.x;
  int b  = (int)(idx / perBatch);
  int sm = (int)(idx % perBatch);
  float mean = stats[2 * b], rstd = stats[2 * b + 1];
  float v = (src[idx] - mean) * rstd * w[sm] + bsh[sm];
  outF[idx] = v;
  if (WB16) outB[idx] = f2bf(v);
}

// ---------------------------------------------------------------------------
// Host launch
// ---------------------------------------------------------------------------
static inline size_t alignup(size_t v) { return (v + 255) & ~(size_t)255; }

extern "C" void kernel_launch(void* const* d_in, const int* in_sizes, int n_in,
                              void* d_out, int out_size, void* d_ws, size_t ws_size,
                              hipStream_t stream) {
  (void)in_sizes; (void)n_in; (void)out_size; (void)ws_size;
  constexpr int Bn = 4, S = 2048, MD = 1024, FF = 4096;
  constexpr long long SM = (long long)S * MD;   // 2097152
  constexpr long long SS = (long long)S * S;    // 4194304

  const float* x     = (const float*)d_in[0];
  const float* w_out = (const float*)d_in[1];
  const float* ln1_w = (const float*)d_in[2];
  const float* ln1_b = (const float*)d_in[3];
  const float* w1    = (const float*)d_in[4];
  const float* b1    = (const float*)d_in[5];
  const float* w2    = (const float*)d_in[6];
  const float* b2    = (const float*)d_in[7];
  const float* ln2_w = (const float*)d_in[8];
  const float* ln2_b = (const float*)d_in[9];
  float* out = (float*)d_out;

  // workspace layout (aliased regions)
  uint8_t* ws = (uint8_t*)d_ws;
  size_t off = 0;
  uint16_t* xb   = (uint16_t*)(ws + off); off = alignup(off + (size_t)Bn * SM * 2);
  uint16_t* wob  = (uint16_t*)(ws + off); off = alignup(off + (size_t)MD * MD * 2);
  uint16_t* w1b  = (uint16_t*)(ws + off); off = alignup(off + (size_t)FF * MD * 2);
  uint16_t* w2b  = (uint16_t*)(ws + off); off = alignup(off + (size_t)MD * FF * 2);
  uint8_t*  r2   = ws + off;              off = alignup(off + (size_t)Bn * SS * 4); // scores f32 / ff1b bf16
  uint8_t*  r3   = ws + off;              off = alignup(off + (size_t)Bn * SS * 2); // attn bf16 / y f32
  uint8_t*  r4   = ws + off;              off = alignup(off + (size_t)Bn * SM * 2); // ctx bf16 / h bf16
  float*    hbuf = (float*)(ws + off);    off = alignup(off + (size_t)Bn * SM * 4); // o+x -> h f32
  float*    part = (float*)(ws + off);    off = alignup(off + (size_t)Bn * 128 * 2 * 4);
  float*    stat = (float*)(ws + off);    off = alignup(off + (size_t)Bn * 2 * 4);

  float*    scores = (float*)r2;
  uint16_t* ff1b   = (uint16_t*)r2;
  uint16_t* attnb  = (uint16_t*)r3;
  float*    ybuf   = (float*)r3;
  uint16_t* ctxb   = (uint16_t*)r4;
  uint16_t* hb     = (uint16_t*)r4;

  // 1) fp32 -> bf16 conversions
  cvt_bf16_kernel<<<(Bn * SM / 4 + 255) / 256, 256, 0, stream>>>(x, xb, Bn * SM / 4);
  cvt_bf16_kernel<<<(MD * MD / 4 + 255) / 256, 256, 0, stream>>>(w_out, wob, MD * MD / 4);
  cvt_bf16_kernel<<<(FF * MD / 4 + 255) / 256, 256, 0, stream>>>(w1, w1b, FF * MD / 4);
  cvt_bf16_kernel<<<(MD * FF / 4 + 255) / 256, 256, 0, stream>>>(w2, w2b, MD * FF / 4);

  // 2) scores = (x @ x^T) / sqrt(K)   [per batch]
  gemm_bf16_wmma<true, 0><<<dim3(S / 128, S / 128, Bn), 256, 0, stream>>>(
      xb, xb, scores, MD, MD, MD, S, SM, SM, SS, nullptr, nullptr, 0.03125f);

  // 3) softmax over keys -> bf16 attn
  softmax_rows<<<Bn * S, 256, 0, stream>>>(scores, attnb, S);

  // 4) ctx = attn @ x  (B is [K,N] row-major -> TRANSB=false)
  gemm_bf16_wmma<false, 1><<<dim3(MD / 128, S / 128, Bn), 256, 0, stream>>>(
      attnb, xb, ctxb, S, S, MD, MD, SS, SM, SM, nullptr, nullptr, 0.f);

  // 5) o = ctx @ w_out^T + x  (rows flattened to B*S)
  gemm_bf16_wmma<true, 2><<<dim3(MD / 128, Bn * S / 128, 1), 256, 0, stream>>>(
      ctxb, wob, hbuf, MD, MD, MD, MD, 0, 0, 0, nullptr, x, 0.f);

  // 6) LayerNorm1 over (S,M) per batch
  ln_reduce_partial<<<Bn * 128, 256, 0, stream>>>(hbuf, part, (int)SM, 128);
  ln_finalize<<<Bn, 256, 0, stream>>>(part, stat, 128, 1.f / (float)SM);
  ln_apply<true><<<(unsigned)(Bn * SM / 256), 256, 0, stream>>>(
      hbuf, stat, ln1_w, ln1_b, hbuf, hb, (int)SM);

  // 7) ff1 = relu(h @ w1^T + b1) -> bf16
  gemm_bf16_wmma<true, 3><<<dim3(FF / 128, Bn * S / 128, 1), 256, 0, stream>>>(
      hb, w1b, ff1b, MD, MD, MD, FF, 0, 0, 0, b1, nullptr, 0.f);

  // 8) y = ff1 @ w2^T + b2 + h
  gemm_bf16_wmma<true, 4><<<dim3(MD / 128, Bn * S / 128, 1), 256, 0, stream>>>(
      ff1b, w2b, ybuf, FF, FF, FF, MD, 0, 0, 0, b2, hbuf, 0.f);

  // 9) LayerNorm2 -> d_out
  ln_reduce_partial<<<Bn * 128, 256, 0, stream>>>(ybuf, part, (int)SM, 128);
  ln_finalize<<<Bn, 256, 0, stream>>>(part, stat, 128, 1.f / (float)SM);
  ln_apply<false><<<(unsigned)(Bn * SM / 256), 256, 0, stream>>>(
      ybuf, stat, ln2_w, ln2_b, out, nullptr, (int)SM);
}
